// MultiheadSelfAttention_26439818674547
// MI455X (gfx1250) — matrix operand order repaired and checked by reading, not compiled, not run on hardware
//
#include <hip/hip_runtime.h>
#include <hip/hip_bf16.h>

#define DEV __device__ __forceinline__

typedef __attribute__((ext_vector_type(16))) __bf16 v16bf;
typedef __attribute__((ext_vector_type(8)))  __bf16 v8bf;
typedef __attribute__((ext_vector_type(8)))  float  v8f;
typedef __attribute__((ext_vector_type(4)))  unsigned int v4u;
typedef __attribute__((ext_vector_type(8)))  int v8i;
typedef __attribute__((ext_vector_type(4)))  int v4i;

constexpr int S  = 2048;
constexpr int Bz = 4;
constexpr int D  = 1024;
constexpr int H  = 16;
constexpr int E  = 64;
constexpr int M  = Bz * S;          // 8192 projection rows
constexpr float SCALE = 0.125f;     // 1/sqrt(E)

// ---------------- WMMA fragment helpers (layouts per CDNA5 ISA 7.12.2) -----

DEV int lane_id() { return (int)(threadIdx.x & 31); }

// A-matrix 16x32 bf16 fragment from LDS tile stored [m][k] (row stride*2 bytes
// must be a multiple of 16). Lane L: m=L%16, K halves at 8*(L/16) and
// 16+8*(L/16) -> two 16-byte ds_load_b128.
DEV v16bf lds_a_frag(const __bf16* base, int stride) {
  int lane = lane_id();
  const __bf16* p = base + (lane & 15) * stride + ((lane >> 4) * 8);
  v8bf lo = *(const v8bf*)(p);
  v8bf hi = *(const v8bf*)(p + 16);
  v16bf r;
#pragma unroll
  for (int i = 0; i < 8; ++i) { r[i] = lo[i]; r[i + 8] = hi[i]; }
  return r;
}

// B-matrix 32x16 bf16 fragment from LDS tile stored TRANSPOSED [n][k].
// Lane L: n=L%16, K halves at 8*(L/16) and 16+8*(L/16) -> two 16-byte loads.
DEV v16bf lds_b_frag(const __bf16* baseT, int stride) {
  int lane = lane_id();
  const __bf16* p = baseT + (lane & 15) * stride + ((lane >> 4) * 8);
  v8bf lo = *(const v8bf*)(p);
  v8bf hi = *(const v8bf*)(p + 16);
  v16bf r;
#pragma unroll
  for (int i = 0; i < 8; ++i) { r[i] = lo[i]; r[i + 8] = hi[i]; }
  return r;
}

DEV v8f wmma_bf16(v16bf a, v16bf b, v8f c) {
  return __builtin_amdgcn_wmma_f32_16x16x32_bf16(
      /*neg_a=*/false, a, /*neg_b=*/false, b,
      /*c_mod=*/(short)0, c, /*reuse_a=*/false, /*reuse_b=*/false);
}

// ---------------- precision / layout conversion kernels --------------------

// x [S,B,D] f32 -> xb [B,S,D] bf16 (batch-first)
__global__ void k_convert_x(const float* __restrict__ in, __bf16* __restrict__ out) {
  int idx = blockIdx.x * 256 + threadIdx.x;   // exact S*B*D
  int d = idx % D;
  int t = idx / D;          // t = s*B + b
  int b = t % Bz;
  int s = t / Bz;
  out[((size_t)(b * S + s)) * D + d] = (__bf16)in[idx];
}

// W [H,D,E] f32 -> wb [D, H*E] bf16 so the projection is a plain GEMM
__global__ void k_convert_wqkv(const float* __restrict__ w, __bf16* __restrict__ out) {
  int idx = blockIdx.x * 256 + threadIdx.x;   // exact H*D*E
  int h = idx / (D * E);
  int r = idx % (D * E);
  int d = r / E;
  int e = r % E;
  out[(size_t)d * D + h * E + e] = (__bf16)w[idx];
}

// Wo [D,D] f32 -> bf16 (already [k][n])
__global__ void k_convert_wo(const float* __restrict__ w, __bf16* __restrict__ out) {
  int idx = blockIdx.x * 256 + threadIdx.x;   // exact D*D
  out[idx] = (__bf16)w[idx];
}

// ---------------- bf16 WMMA GEMM: C[M,D] = A[M,D] @ Bw[D,D] + bias ---------
// A tile staged by the Tensor Data Mover (TDM); B tile staged manually
// (transposed). OUT_MODE 0: bf16 [B,H,S,E]; OUT_MODE 1: f32 row-major [M,D].
template <int OUT_MODE>
__global__ __launch_bounds__(256) void k_gemm(const __bf16* __restrict__ A,
                                              const __bf16* __restrict__ Bw,
                                              const float* __restrict__ bias,
                                              void* __restrict__ outp) {
  constexpr int BM = 128, BN = 64, BK = 64;
  constexpr int KS = BK + 8;                    // 72 elems = 144 B rows
  __shared__ __align__(16) __bf16 a_lds[BM * KS];   // 18.0 KB (TDM dest)
  __shared__ __align__(16) __bf16 bT_lds[BN * KS];  //  9.0 KB

  const int ntn = D / BN;                       // 16
  int tn = blockIdx.x % ntn, tm = blockIdx.x / ntn;
  int m0 = tm * BM, n0 = tn * BN;
  int tid = threadIdx.x;
  int w = tid >> 5, lane = tid & 31;
  int wm = w >> 1, wn = w & 1;                  // 4x2 wave grid, 32x32 per wave

  // LDS byte offset of a_lds (flat LDS aperture truncates to addr[31:0])
  unsigned lds_a_addr = (unsigned)(size_t)(&a_lds[0]);

  v8f acc[2][2] = {};

  for (int kb = 0; kb < D; kb += BK) {
    __syncthreads();                            // all readers done with tiles

    if (w == 0) {                               // wave-uniform: single TDM issue
      // ---- Tensor DMA descriptor (CDNA5 ISA ch.8), 2D tile 128x64 bf16 ----
      unsigned long long ga =
          (unsigned long long)(size_t)(A + (size_t)m0 * D + kb);
      v4u g0;
      g0[0] = 1u;                                         // count=1, user mode
      g0[1] = lds_a_addr;                                 // lds_addr
      g0[2] = (unsigned)(ga & 0xffffffffu);               // global_addr[31:0]
      g0[3] = (unsigned)((ga >> 32) & 0x01ffffffu)        // global_addr[56:32]
              | (2u << 30);                               // type = 2 ("image")
      v8i g1;
      g1[0] = (1 << 16)      // data_size = 1 -> 2 bytes
            | (1 << 20)      // pad_enable (LDS row padding)
            | (4 << 22)      // pad_interval = 4 -> every 32 DWORDs (=128B row)
            | (3 << 25);     // pad_amount  = 3 -> 4 DWORDs (=16B) => 72-el rows
      g1[1] = (int)(((unsigned)D & 0xffffu) << 16);       // tensor_dim0 lo16
      g1[2] = (int)((((unsigned)D >> 16) & 0xffffu)       // tensor_dim0 hi16
              | (((unsigned)M & 0xffffu) << 16));         // tensor_dim1 lo16
      g1[3] = (int)((((unsigned)M >> 16) & 0xffffu)       // tensor_dim1 hi16
              | (64u << 16));                             // tile_dim0 = 64
      g1[4] = 128;                                        // tile_dim1 = 128
      g1[5] = D;                                          // dim0_stride lo32
      g1[6] = 0;
      g1[7] = 0;
      v4i gz4 = {};
      v8i gz8 = {};
      __builtin_amdgcn_tensor_load_to_lds(g0, g1, gz4, gz4, gz8, 0);
    }

    { // B tile 64x64 -> stored transposed [n][k] (TDM cannot transpose)
      int k = tid >> 2, nn = (tid & 3) * 16;
      const __bf16* src = Bw + (size_t)(kb + k) * D + n0 + nn;
      v8bf y0 = *(const v8bf*)(src);
      v8bf y1 = *(const v8bf*)(src + 8);
#pragma unroll
      for (int i = 0; i < 8; ++i) {
        bT_lds[(nn + i) * KS + k]     = y0[i];
        bT_lds[(nn + 8 + i) * KS + k] = y1[i];
      }
      if (kb + BK < D) __builtin_prefetch(src + (size_t)BK * D, 0, 3); // next B
    }

    if (w == 0) __builtin_amdgcn_s_wait_tensorcnt(0);     // TDM complete
    __syncthreads();                                      // publish tiles

#pragma unroll
    for (int ks = 0; ks < 2; ++ks) {            // two K=32 sub-steps
      v16bf af[2], bfm[2];
#pragma unroll
      for (int i = 0; i < 2; ++i)
        af[i] = lds_a_frag(&a_lds[(wm * 32 + i * 16) * KS + 32 * ks], KS);
#pragma unroll
      for (int i = 0; i < 2; ++i)
        bfm[i] = lds_b_frag(&bT_lds[(wn * 32 + i * 16) * KS + 32 * ks], KS);
#pragma unroll
      for (int mi = 0; mi < 2; ++mi)
#pragma unroll
        for (int ni = 0; ni < 2; ++ni)
          acc[mi][ni] = wmma_bf16(af[mi], bfm[ni], acc[mi][ni]);
    }
  }

  // epilogue: C layout VGPR r -> row r+8*(lane/16), col lane%16
  int hi = lane >> 4, col = lane & 15;
#pragma unroll
  for (int mi = 0; mi < 2; ++mi)
#pragma unroll
    for (int ni = 0; ni < 2; ++ni)
#pragma unroll
      for (int r = 0; r < 8; ++r) {
        int m = m0 + wm * 32 + mi * 16 + r + 8 * hi;
        int n = n0 + wn * 32 + ni * 16 + col;
        float v = acc[mi][ni][r] + bias[n];
        if constexpr (OUT_MODE == 0) {
          int b = m >> 11, s = m & (S - 1);       // S = 2048
          int h = n >> 6,  e = n & (E - 1);       // E = 64
          ((__bf16*)outp)[(((size_t)(b * H + h) * S + s) * E) + e] = (__bf16)v;
        } else {
          ((float*)outp)[(size_t)m * D + n] = v;
        }
      }
}

// ---------------- mask-as-matmul == reverse exclusive suffix sum -----------
// kmask[bh, t, e] = sum_{s > t} k[bh, s, e]   (fp32 accumulate)
__global__ void k_kmask(const __bf16* __restrict__ k, __bf16* __restrict__ km,
                        const int* __restrict__ use_mask) {
  int tid = blockIdx.x * 256 + threadIdx.x;   // exact B*H*E = 4096
  int e  = tid & (E - 1);
  int bh = tid >> 6;
  const __bf16* kp = k  + (size_t)bh * S * E + e;
  __bf16*      kmp = km + (size_t)bh * S * E + e;
  if (use_mask[0] == 0) {
    for (int t = 0; t < S; ++t) kmp[(size_t)t * E] = kp[(size_t)t * E];
    return;
  }
  float acc = 0.f;
  for (int t = S - 1; t >= 0; --t) {
    kmp[(size_t)t * E] = (__bf16)acc;       // exclusive: s > t
    acc += (float)kp[(size_t)t * E];
  }
}

// ---------------- fused flash attention (online softmax) -------------------
// grid = B*H * (S/128); 8 waves, 16 query rows each; key blocks of 64.
__global__ __launch_bounds__(256) void k_attn(const __bf16* __restrict__ Q,
                                              const __bf16* __restrict__ Km,
                                              const __bf16* __restrict__ V,
                                              __bf16* __restrict__ Out) {
  constexpr int BQ = 128, BKEY = 64;
  constexpr int QS = E + 8;        // 72 elems = 144 B rows (over e)
  constexpr int TS = BKEY + 8;     // 72 elems rows of vT   (over t)
  constexpr int PS = BKEY + 8;     // 72 elems rows of P    (over key col)
  __shared__ __align__(16) __bf16 q_lds[BQ * QS];        // 18.0 KB
  __shared__ __align__(16) __bf16 km_lds[BKEY * QS];     //  9.0 KB
  __shared__ __align__(16) __bf16 vT_lds[E * TS];        //  9.0 KB
  __shared__ __align__(16) __bf16 p_lds[8 * 16 * PS];    // 18.0 KB (per-wave P)

  const int nqb = S / BQ;                     // 16
  int qb = blockIdx.x % nqb;
  int bh = blockIdx.x / nqb;
  int b = bh >> 4, h = bh & 15;
  int tid = threadIdx.x, w = tid >> 5, lane = tid & 31;
  int hi = lane >> 4, col = lane & 15;

  const __bf16* Qb = Q  + (size_t)bh * S * E;
  const __bf16* Kb = Km + (size_t)bh * S * E;
  const __bf16* Vb = V  + (size_t)bh * S * E;

  { // stage the whole 128x64 Q block once
    int q0 = qb * BQ;
#pragma unroll
    for (int i = 0; i < 4; ++i) {
      int idx = (tid + i * 256) * 8;          // 8192 elems
      int r = idx >> 6, e = idx & 63;
      v8bf x = *(const v8bf*)(Qb + (size_t)(q0 + r) * E + e);
      *(v8bf*)(&q_lds[r * QS + e]) = x;
    }
  }
  __syncthreads();

  v16bf qf[2];                                // A-frags for e in [0,32),[32,64)
#pragma unroll
  for (int j = 0; j < 2; ++j) qf[j] = lds_a_frag(&q_lds[(w * 16) * QS + 32 * j], QS);

  v8f o[4] = {};
  float mrow[8], lrow[8];
#pragma unroll
  for (int r = 0; r < 8; ++r) { mrow[r] = -1e30f; lrow[r] = 0.f; }

  for (int t0 = 0; t0 < S; t0 += BKEY) {
    __syncthreads();                          // protect km/vT from prior readers
    { // stage Km [64][64] natural and V transposed [e][t]; 16 elems/thread each
      int t = tid >> 2, e = (tid & 3) * 16;
      const __bf16* ksrc = Kb + (size_t)(t0 + t) * E + e;
      v8bf x0 = *(const v8bf*)(ksrc);
      v8bf x1 = *(const v8bf*)(ksrc + 8);
      *(v8bf*)(&km_lds[t * QS + e])     = x0;
      *(v8bf*)(&km_lds[t * QS + e + 8]) = x1;
      const __bf16* vsrc = Vb + (size_t)(t0 + t) * E + e;
      v8bf y0 = *(const v8bf*)(vsrc);
      v8bf y1 = *(const v8bf*)(vsrc + 8);
#pragma unroll
      for (int i = 0; i < 8; ++i) {
        vT_lds[(e + i) * TS + t]     = y0[i];
        vT_lds[(e + 8 + i) * TS + t] = y1[i];
      }
    }
    __syncthreads();

    // scores tile 16x64 = four n-tiles; K-dim = E via 2 sub-steps
    v8f sc[4] = {};
#pragma unroll
    for (int ks = 0; ks < 2; ++ks)
#pragma unroll
      for (int nt = 0; nt < 4; ++nt) {
        v16bf bk = lds_b_frag(&km_lds[(nt * 16) * QS + 32 * ks], QS);
        sc[nt] = wmma_bf16(qf[ks], bk, sc[nt]);
      }

    // online softmax: stats per row live per C-VGPR index r (row r+8*hi)
    float alpha[8];
    __bf16* pw = &p_lds[w * 16 * PS];
#pragma unroll
    for (int r = 0; r < 8; ++r) {
      float s0 = sc[0][r] * SCALE, s1 = sc[1][r] * SCALE;
      float s2 = sc[2][r] * SCALE, s3 = sc[3][r] * SCALE;
      float bm = fmaxf(fmaxf(s0, s1), fmaxf(s2, s3));
      bm = fmaxf(bm, __shfl_xor(bm, 1, 32));
      bm = fmaxf(bm, __shfl_xor(bm, 2, 32));
      bm = fmaxf(bm, __shfl_xor(bm, 4, 32));
      bm = fmaxf(bm, __shfl_xor(bm, 8, 32));   // stays inside 16-lane half
      float mn = fmaxf(mrow[r], bm);
      float a  = __expf(mrow[r] - mn);
      float p0 = __expf(s0 - mn), p1 = __expf(s1 - mn);
      float p2 = __expf(s2 - mn), p3 = __expf(s3 - mn);
      float rs = (p0 + p1) + (p2 + p3);
      rs += __shfl_xor(rs, 1, 32);
      rs += __shfl_xor(rs, 2, 32);
      rs += __shfl_xor(rs, 4, 32);
      rs += __shfl_xor(rs, 8, 32);
      lrow[r] = lrow[r] * a + rs;
      mrow[r] = mn;
      alpha[r] = a;
      int row = r + 8 * hi;
      pw[row * PS + col]      = (__bf16)p0;    // P into per-wave LDS, [m][k]
      pw[row * PS + 16 + col] = (__bf16)p1;
      pw[row * PS + 32 + col] = (__bf16)p2;
      pw[row * PS + 48 + col] = (__bf16)p3;
    }
#pragma unroll
    for (int j = 0; j < 4; ++j)
#pragma unroll
      for (int r = 0; r < 8; ++r) o[j][r] *= alpha[r];
    __syncthreads();                          // order P writes vs A-frag reads

    // O += P(16x64) @ V(64x64)
    v16bf pf0 = lds_a_frag(pw, PS);           // keys 0..31 of block
    v16bf pf1 = lds_a_frag(pw + 32, PS);      // keys 32..63
#pragma unroll
    for (int nt = 0; nt < 4; ++nt) {
      v16bf vf0 = lds_b_frag(&vT_lds[(16 * nt) * TS], TS);
      v16bf vf1 = lds_b_frag(&vT_lds[(16 * nt) * TS + 32], TS);
      o[nt] = wmma_bf16(pf0, vf0, o[nt]);
      o[nt] = wmma_bf16(pf1, vf1, o[nt]);
    }
  }

  // normalize and store head-concat layout [B,S,D], d = h*E + e
#pragma unroll
  for (int r = 0; r < 8; ++r) lrow[r] = 1.f / lrow[r];
#pragma unroll
  for (int j = 0; j < 4; ++j)
#pragma unroll
    for (int r = 0; r < 8; ++r) {
      int srow = qb * BQ + w * 16 + r + 8 * hi;
      int d = h * E + 16 * j + col;
      Out[((size_t)(b * S + srow)) * D + d] = (__bf16)(o[j][r] * lrow[r]);
    }
}

// ---------------- launcher --------------------------------------------------

extern "C" void kernel_launch(void* const* d_in, const int* in_sizes, int n_in,
                              void* d_out, int out_size, void* d_ws, size_t ws_size,
                              hipStream_t stream) {
  (void)in_sizes; (void)n_in; (void)out_size; (void)ws_size;
  const float* x  = (const float*)d_in[0];
  const float* Wq = (const float*)d_in[1];
  const float* bq = (const float*)d_in[2];
  const float* Wk = (const float*)d_in[3];
  const float* bk = (const float*)d_in[4];
  const float* Wv = (const float*)d_in[5];
  const float* bv = (const float*)d_in[6];
  const float* Wo = (const float*)d_in[7];
  const float* bo = (const float*)d_in[8];
  const int* use_mask = (const int*)d_in[10];   // [9] = mask_matrix (replaced by scan)
  float* out = (float*)d_out;

  char* ws = (char*)d_ws;
  size_t off = 0;
  auto alloc = [&](size_t bytes) {
    char* p = ws + off;
    off += (bytes + 255) & ~(size_t)255;
    return p;
  };
  __bf16* xb    = (__bf16*)alloc((size_t)M * D * 2);   // 16 MB
  __bf16* wqb   = (__bf16*)alloc((size_t)D * D * 2);   //  2 MB
  __bf16* wkb   = (__bf16*)alloc((size_t)D * D * 2);
  __bf16* wvb   = (__bf16*)alloc((size_t)D * D * 2);
  __bf16* wob   = (__bf16*)alloc((size_t)D * D * 2);
  __bf16* qb    = (__bf16*)alloc((size_t)M * D * 2);   // [B,H,S,E]
  __bf16* kbuf  = (__bf16*)alloc((size_t)M * D * 2);
  __bf16* vb    = (__bf16*)alloc((size_t)M * D * 2);
  __bf16* kmb   = (__bf16*)alloc((size_t)M * D * 2);
  __bf16* attnb = (__bf16*)alloc((size_t)M * D * 2);   // [B,S,D] concat

  k_convert_x<<<(S * Bz * D) / 256, 256, 0, stream>>>(x, xb);
  k_convert_wqkv<<<(H * D * E) / 256, 256, 0, stream>>>(Wq, wqb);
  k_convert_wqkv<<<(H * D * E) / 256, 256, 0, stream>>>(Wk, wkb);
  k_convert_wqkv<<<(H * D * E) / 256, 256, 0, stream>>>(Wv, wvb);
  k_convert_wo<<<(D * D) / 256, 256, 0, stream>>>(Wo, wob);

  dim3 gGemm((M / 128) * (D / 64));             // 1024 workgroups
  k_gemm<0><<<gGemm, 256, 0, stream>>>(xb, wqb, bq, qb);
  k_gemm<0><<<gGemm, 256, 0, stream>>>(xb, wkb, bk, kbuf);
  k_gemm<0><<<gGemm, 256, 0, stream>>>(xb, wvb, bv, vb);

  k_kmask<<<(Bz * H * E) / 256, 256, 0, stream>>>(kbuf, kmb, use_mask);

  k_attn<<<Bz * H * (S / 128), 256, 0, stream>>>(qb, kmb, vb, attnb);

  k_gemm<1><<<gGemm, 256, 0, stream>>>(attnb, wob, bo, out);
}
